// YOLOV3Head_41669772705903
// MI455X (gfx1250) — matrix operand order repaired and checked by reading, not compiled, not run on hardware
//
#include <hip/hip_runtime.h>

// ---------------------------------------------------------------------------
// YOLOv3 head for MI455X (gfx1250): bf16 WMMA implicit-GEMM conv + pred GEMM,
// decode, per-image sort + greedy NMS.  wave32 / WMMA 16x16x32 bf16 -> f32.
// ---------------------------------------------------------------------------

typedef __bf16 bf16;
typedef bf16  bf16x8  __attribute__((ext_vector_type(8)));
typedef bf16  bf16x16 __attribute__((ext_vector_type(16)));
typedef float f32x8   __attribute__((ext_vector_type(8)));

#define NUM_CLS   80
#define ATTRIB    85
#define NTOT      22743      // 1083 + 4332 + 17328 candidates per image
#define SORTN     32768      // padded pow2 for bitonic sort
#define CONF_THRv 0.005f
#define IOU_THRv  0.45f
#define BN_EPSv   1e-5f

__device__ __forceinline__ float sigm(float x) { return 1.f / (1.f + __expf(-x)); }

__device__ __forceinline__ bf16x16 frag_combine(bf16x8 lo, bf16x8 hi) {
  bf16x16 f;
#pragma unroll
  for (int j = 0; j < 8; ++j) { f[j] = lo[j]; f[8 + j] = hi[j]; }
  return f;
}

// --------------------------- weight prep kernels ---------------------------

// wb [Cout][Cin][3][3] f32 -> wA [Cout][K=9*Cin] bf16, K index = (r*3+s)*Cin+c,
// BN scale folded in.
__global__ void prep_wb_k(const float* __restrict__ wb, const float* __restrict__ gamma,
                          const float* __restrict__ var, bf16* __restrict__ wA,
                          int Cout, int Cin) {
  int K = Cin * 9;
  long gid = (long)blockIdx.x * blockDim.x + threadIdx.x;
  if (gid >= (long)Cout * K) return;
  int o  = (int)(gid / K);
  int k  = (int)(gid - (long)o * K);
  int rs = k / Cin;
  int c  = k - rs * Cin;
  float scale = gamma[o] * rsqrtf(var[o] + BN_EPSv);
  float w = wb[((long)o * Cin + c) * 9 + rs];
  wA[gid] = (bf16)(w * scale);
}

__global__ void prep_bias_k(const float* __restrict__ gamma, const float* __restrict__ beta,
                            const float* __restrict__ mean,  const float* __restrict__ var,
                            float* __restrict__ biasb, int Cout) {
  int o = blockIdx.x * blockDim.x + threadIdx.x;
  if (o >= Cout) return;
  float scale = gamma[o] * rsqrtf(var[o] + BN_EPSv);
  biasb[o] = beta[o] - mean[o] * scale;
}

// wp [255][Cout] f32 -> wpbf [256][Cout] bf16 (row 255 zero-padded)
__global__ void prep_wp_k(const float* __restrict__ wp, bf16* __restrict__ wpbf, int Cout) {
  long gid = (long)blockIdx.x * blockDim.x + threadIdx.x;
  if (gid >= (long)256 * Cout) return;
  int m = (int)(gid / Cout);
  wpbf[gid] = (m < 255) ? (bf16)wp[gid] : (bf16)0.f;
}

// ------------------- 3x3 conv + BN + LeakyReLU (WMMA GEMM) -----------------
// GEMM: M=Cout, N=16*H*W, K=9*Cin.  Block tile 128x64x32, 8 waves, each wave
// computes 2x2 WMMA 16x16 tiles.  K loop split as (tap rs) x (channel chunk)
// so all im2col divisions are hoisted out of the hot loop.  Out-of-bounds
// taps load from a safe in-bounds address (unpredicated loads) and are zeroed
// by a multiply -- avoids per-element exec-mask load predication.
// Output y (bf16) stored transposed: [N][Cout].
__global__ __launch_bounds__(256)
void conv3x3_wmma_k(const float* __restrict__ x, const bf16* __restrict__ wA,
                    const float* __restrict__ biasb, bf16* __restrict__ y,
                    int Cin, int H, int W, int N, int Cout) {
  __shared__ bf16 As[128 * 40];   // 128 rows x 32 (pad to 40 halves / 80B stride)
  __shared__ bf16 Bs[64 * 40];    // 64 n-cols x 32 k (pad to 40)

  const int K    = Cin * 9;
  const int HW   = H * W;
  const int tid  = threadIdx.x;
  const int lane = tid & 31;
  const int wave = tid >> 5;
  const int wm   = wave & 3;          // 4 waves along M (4*32 = 128)
  const int wn   = wave >> 2;         // 2 waves along N (2*32 = 64)
  const int mBase = blockIdx.y * 128;
  const int nBase = blockIdx.x * 64;

  // B staging mapping: each thread loads 8 consecutive input channels for one n
  const int bn = tid & 63;
  const int bk = (tid >> 6) * 8;
  const int n  = nBase + bn;
  int b = 0, h = 0, w = 0;
  const bool nvalid = (n < N);
  if (nvalid) { b = n / HW; int hw2 = n - b * HW; h = hw2 / W; w = hw2 - h * W; }

  // A staging mapping: 16 halves per thread, pointer advanced incrementally
  const int ar = tid >> 1;
  const int ac = (tid & 1) * 16;
  const uint4* gA = (const uint4*)(wA + (long)(mBase + ar) * K + ac);

  // fragment load addresses (constant across K loop)
  const int mrow0 = wm * 32 + (lane & 15);
  const int koff  = (lane >> 4) * 16;  // A: lanes 0-15 -> K 0..15; 16-31 -> K 16..31
  const int ncol0 = wn * 32 + (lane & 15);
  const int kb    = (lane >> 4) * 16;  // B: lanes 0-15 -> K 0..15; 16-31 -> K 16..31

  f32x8 acc[2][2];
#pragma unroll
  for (int i = 0; i < 2; ++i)
#pragma unroll
    for (int j = 0; j < 2; ++j)
#pragma unroll
      for (int e = 0; e < 8; ++e) acc[i][j][e] = 0.f;

  for (int rs = 0; rs < 9; ++rs) {
    const int dr = rs / 3 - 1, dsd = rs - (rs / 3) * 3 - 1;
    const int ih = h + dr, iw = w + dsd;
    const bool ok = nvalid && ((unsigned)ih < (unsigned)H) && ((unsigned)iw < (unsigned)W);
    const float mask = ok ? 1.f : 0.f;
    // safe base: real tap address when ok, else an always-in-bounds dummy
    // (x + bk*HW stays in-bounds for the whole channel walk below)
    const float* p = ok ? (x + (((long)b * Cin + bk) * H + ih) * W + iw)
                        : (x + (long)bk * HW);

    for (int c0 = 0; c0 < Cin; c0 += 32) {
      // stage A (weights, bf16 row-major [M][K]); k advances contiguously
      uint4 v0 = gA[0];
      uint4 v1 = gA[1];
      gA += 4;
      *(uint4*)(&As[ar * 40 + ac])     = v0;
      *(uint4*)(&As[ar * 40 + ac + 8]) = v1;

      // stage B (im2col of x): unpredicated loads, multiply-mask for padding
      float fv[8];
#pragma unroll
      for (int j = 0; j < 8; ++j) fv[j] = p[(long)j * HW];
      p += (long)32 * HW;
      bf16x8 vv;
#pragma unroll
      for (int j = 0; j < 8; ++j) vv[j] = (bf16)(fv[j] * mask);
      *(bf16x8*)(&Bs[bn * 40 + bk]) = vv;
      __syncthreads();

      bf16x16 aF[2], bF[2];
#pragma unroll
      for (int mt = 0; mt < 2; ++mt) {
        int row = mrow0 + mt * 16;
        bf16x8 lo = *(const bf16x8*)(&As[row * 40 + koff]);
        bf16x8 hi = *(const bf16x8*)(&As[row * 40 + koff + 8]);
        aF[mt] = frag_combine(lo, hi);
      }
#pragma unroll
      for (int nt = 0; nt < 2; ++nt) {
        int col = ncol0 + nt * 16;
        bf16x8 lo = *(const bf16x8*)(&Bs[col * 40 + kb]);
        bf16x8 hi = *(const bf16x8*)(&Bs[col * 40 + kb + 8]);
        bF[nt] = frag_combine(lo, hi);
      }
#pragma unroll
      for (int mt = 0; mt < 2; ++mt)
#pragma unroll
        for (int nt = 0; nt < 2; ++nt)
          acc[mt][nt] = __builtin_amdgcn_wmma_f32_16x16x32_bf16(
              false, aF[mt], false, bF[nt], (short)0, acc[mt][nt], false, false);
      __syncthreads();
    }
  }

  // epilogue: bias + LeakyReLU, store bf16 y[N][Cout] (8 contiguous m per store)
#pragma unroll
  for (int mt = 0; mt < 2; ++mt) {
#pragma unroll
    for (int nt = 0; nt < 2; ++nt) {
      int ncol = nBase + wn * 32 + nt * 16 + (lane & 15);
      if (ncol >= N) continue;
      int m0 = mBase + wm * 32 + mt * 16 + ((lane >> 4) << 3);
      bf16x8 outp;
#pragma unroll
      for (int r = 0; r < 8; ++r) {
        float v = acc[mt][nt][r] + biasb[m0 + r];
        v = (v >= 0.f) ? v : 0.1f * v;
        outp[r] = (bf16)v;
      }
      *(bf16x8*)(&y[(long)ncol * Cout + m0]) = outp;
    }
  }
}

// --------------------------- 1x1 pred conv (WMMA) --------------------------
// GEMM: M=256(255 real), N=16*H*W, K=Cout.  A = wpbf [256][K] bf16,
// B = y [N][K] bf16 (contiguous loads).  Output pm f32 [N][256].
// Rows with n >= N clamp to row 0 (their outputs are never stored).
__global__ __launch_bounds__(256)
void pred_wmma_k(const bf16* __restrict__ yin, const bf16* __restrict__ wpbf,
                 const float* __restrict__ bp, float* __restrict__ pm, int N, int K) {
  __shared__ bf16 As[128 * 40];
  __shared__ bf16 Bs[64 * 40];

  const int tid  = threadIdx.x;
  const int lane = tid & 31;
  const int wave = tid >> 5;
  const int wm   = wave & 3;
  const int wn   = wave >> 2;
  const int mBase = blockIdx.y * 128;
  const int nBase = blockIdx.x * 64;

  // B staging: 4 consecutive threads cover one contiguous 64B run of y[n][*]
  const int bn = tid >> 2;
  const int bk = (tid & 3) * 8;
  const int n  = nBase + bn;
  const long nrow = (n < N) ? (long)n : 0;             // safe row, result unused
  const uint4* gB = (const uint4*)(yin + nrow * K + bk);

  const int ar = tid >> 1;
  const int ac = (tid & 1) * 16;
  const uint4* gA = (const uint4*)(wpbf + (long)(mBase + ar) * K + ac);

  const int mrow0 = wm * 32 + (lane & 15);
  const int koff  = (lane >> 4) * 16;
  const int ncol0 = wn * 32 + (lane & 15);
  const int kb    = (lane >> 4) * 16;

  f32x8 acc[2][2];
#pragma unroll
  for (int i = 0; i < 2; ++i)
#pragma unroll
    for (int j = 0; j < 2; ++j)
#pragma unroll
      for (int e = 0; e < 8; ++e) acc[i][j][e] = 0.f;

  for (int k0 = 0; k0 < K; k0 += 32) {
    uint4 v0 = gA[0];
    uint4 v1 = gA[1];
    gA += 4;
    *(uint4*)(&As[ar * 40 + ac])     = v0;
    *(uint4*)(&As[ar * 40 + ac + 8]) = v1;

    uint4 vb = gB[0];
    gB += 4;
    *(uint4*)(&Bs[bn * 40 + bk]) = vb;
    __syncthreads();

    bf16x16 aF[2], bF[2];
#pragma unroll
    for (int mt = 0; mt < 2; ++mt) {
      int row = mrow0 + mt * 16;
      bf16x8 lo = *(const bf16x8*)(&As[row * 40 + koff]);
      bf16x8 hi = *(const bf16x8*)(&As[row * 40 + koff + 8]);
      aF[mt] = frag_combine(lo, hi);
    }
#pragma unroll
    for (int nt = 0; nt < 2; ++nt) {
      int col = ncol0 + nt * 16;
      bf16x8 lo = *(const bf16x8*)(&Bs[col * 40 + kb]);
      bf16x8 hi = *(const bf16x8*)(&Bs[col * 40 + kb + 8]);
      bF[nt] = frag_combine(lo, hi);
    }
#pragma unroll
    for (int mt = 0; mt < 2; ++mt)
#pragma unroll
      for (int nt = 0; nt < 2; ++nt)
        acc[mt][nt] = __builtin_amdgcn_wmma_f32_16x16x32_bf16(
            false, aF[mt], false, bF[nt], (short)0, acc[mt][nt], false, false);
    __syncthreads();
  }

  // epilogue: pm[N][256], two aligned float4 stores (row 255 is padding)
#pragma unroll
  for (int mt = 0; mt < 2; ++mt) {
#pragma unroll
    for (int nt = 0; nt < 2; ++nt) {
      int ncol = nBase + wn * 32 + nt * 16 + (lane & 15);
      if (ncol >= N) continue;
      int m0 = mBase + wm * 32 + mt * 16 + ((lane >> 4) << 3);
      float o[8];
#pragma unroll
      for (int r = 0; r < 8; ++r) {
        int mi = m0 + r;
        o[r] = acc[mt][nt][r] + bp[mi < 255 ? mi : 254];
      }
      float* dst = pm + (long)ncol * 256 + m0;
      *(float4*)(dst)     = make_float4(o[0], o[1], o[2], o[3]);
      *(float4*)(dst + 4) = make_float4(o[4], o[5], o[6], o[7]);
    }
  }
}

// ------------------------------ box decode ---------------------------------
// pm layout [N][256]: 85 contiguous attributes per (n, anchor)
__global__ void decode_k(const float* __restrict__ pm, float* __restrict__ boxes,
                         float* __restrict__ confb, float* __restrict__ clsb,
                         int* __restrict__ labb, int Hl, int Wl, int N, int strideL,
                         float aw0, float ah0, float aw1, float ah1, float aw2, float ah2,
                         int candOff) {
  long gid = (long)blockIdx.x * blockDim.x + threadIdx.x;
  if (gid >= (long)N * 3) return;
  int a  = (int)(gid % 3);
  int n  = (int)(gid / 3);
  int hw = n % (Hl * Wl);
  int b  = n / (Hl * Wl);
  int h  = hw / Wl, w = hw - h * Wl;

  const float* q = pm + (long)n * 256 + a * ATTRIB;
  float t0 = q[0], t1 = q[1], t2 = q[2], t3 = q[3], t4 = q[4];

  float aw = (a == 0) ? aw0 : ((a == 1) ? aw1 : aw2);
  float ah = (a == 0) ? ah0 : ((a == 1) ? ah1 : ah2);

  float cx = (w + 0.5f) * (float)strideL;
  float cy = (h + 0.5f) * (float)strideL;
  float bx = cx + (sigm(t0) - 0.5f) * (float)strideL;
  float by = cy + (sigm(t1) - 0.5f) * (float)strideL;
  float hwd = 0.5f * aw * __expf(t2);
  float hhd = 0.5f * ah * __expf(t3);

  float best = -3.4e38f; int bi = 0;
#pragma unroll 4
  for (int c = 0; c < NUM_CLS; ++c) {
    float v = q[5 + c];
    if (v > best) { best = v; bi = c; }
  }

  int gi = b * NTOT + candOff + hw * 3 + a;
  boxes[(long)gi * 4 + 0] = bx - hwd;
  boxes[(long)gi * 4 + 1] = by - hhd;
  boxes[(long)gi * 4 + 2] = bx + hwd;
  boxes[(long)gi * 4 + 3] = by + hhd;
  confb[gi] = sigm(t4);
  clsb[gi]  = sigm(best);
  labb[gi]  = bi;
}

// ------------------------ per-image sort (bitonic) -------------------------
__global__ void sort_init_k(const float* __restrict__ confb,
                            unsigned long long* __restrict__ keys) {
  int gid = blockIdx.x * blockDim.x + threadIdx.x;      // 16 * 32768
  int img = gid >> 15;
  int i   = gid & (SORTN - 1);
  float c = 0.f;
  unsigned sec = 0u;                                    // padding sorts last
  if (i < NTOT) {
    c = confb[img * NTOT + i];
    if (c < CONF_THRv) c = 0.f;
    sec = 0x7FFFFFFFu - (unsigned)i;                    // ties: smaller idx first
  }
  keys[gid] = ((unsigned long long)__float_as_uint(c) << 32) | (unsigned long long)sec;
}

__global__ void bitonic_k(unsigned long long* __restrict__ keys, int k, int j) {
  int gid = blockIdx.x * blockDim.x + threadIdx.x;
  int img = gid >> 15;
  int i   = gid & (SORTN - 1);
  int ixj = i ^ j;
  if (ixj <= i) return;
  unsigned long long* kk = keys + (long)img * SORTN;
  unsigned long long a = kk[i], bb = kk[ixj];
  bool up = (i & k) == 0;                               // descending overall
  bool sw = up ? (a < bb) : (a > bb);
  if (sw) { kk[i] = bb; kk[ixj] = a; }
}

// ------------------------------- NMS ---------------------------------------
__global__ __launch_bounds__(1024)
void nms_k(const unsigned long long* __restrict__ keys, const float* __restrict__ boxes,
           const float* __restrict__ clsb, const int* __restrict__ labb,
           float* __restrict__ det, int* __restrict__ outlab) {
  __shared__ float sx1[1024], sy1[1024], sx2[1024], sy2[1024], ssc[1024];
  __shared__ int   slb[1024], sidx[1024];
  __shared__ float skey[1024];
  __shared__ unsigned char skeep[1024];

  const int t = threadIdx.x, img = blockIdx.x;

  float score = -1.f, b0 = 0.f, b1 = 0.f, b2 = 0.f, b3 = 0.f;
  int lab = 0;
  if (t < 1000) {
    unsigned long long kv = keys[(long)img * SORTN + t];
    float c = __uint_as_float((unsigned)(kv >> 32));    // thresholded conf
    unsigned ridx = 0x7FFFFFFFu - (unsigned)(kv & 0xFFFFFFFFu);
    long gi = (long)img * NTOT + (long)ridx;
    score = c * clsb[gi];
    b0 = boxes[gi * 4 + 0]; b1 = boxes[gi * 4 + 1];
    b2 = boxes[gi * 4 + 2]; b3 = boxes[gi * 4 + 3];
    lab = labb[gi];
  }
  skey[t] = score; sidx[t] = t;
  sx1[t] = b0; sy1[t] = b1; sx2[t] = b2; sy2[t] = b3; ssc[t] = score; slb[t] = lab;
  __syncthreads();

  // bitonic sort (descending by score) over 1024 entries in LDS
  for (int k = 2; k <= 1024; k <<= 1) {
    for (int j = k >> 1; j > 0; j >>= 1) {
      int ixj = t ^ j;
      if (ixj > t) {
        float a = skey[t], bb = skey[ixj];
        bool up = (t & k) == 0;
        bool sw = up ? (a < bb) : (a > bb);
        if (sw) {
          skey[t] = bb; skey[ixj] = a;
          int tmp = sidx[t]; sidx[t] = sidx[ixj]; sidx[ixj] = tmp;
        }
      }
      __syncthreads();
    }
  }

  // gather payload into sorted order
  int src = sidx[t];
  float gx1 = sx1[src], gy1 = sy1[src], gx2 = sx2[src], gy2 = sy2[src], gsc = ssc[src];
  int glb = slb[src];
  __syncthreads();
  sx1[t] = gx1; sy1[t] = gy1; sx2[t] = gx2; sy2[t] = gy2; ssc[t] = gsc; slb[t] = glb;
  skeep[t] = (t < 1000 && gsc > 0.f) ? 1 : 0;
  __syncthreads();

  const float mx1 = sx1[t], my1 = sy1[t], mx2 = sx2[t], my2 = sy2[t];
  const float myArea = (mx2 - mx1) * (my2 - my1);

  for (int i = 0; i < 999; ++i) {
    if (skeep[i]) {
      float ix1 = sx1[i], iy1 = sy1[i], ix2 = sx2[i], iy2 = sy2[i];
      if (t > i && skeep[t]) {
        float lt0 = fmaxf(ix1, mx1), lt1 = fmaxf(iy1, my1);
        float rb0 = fminf(ix2, mx2), rb1 = fminf(iy2, my2);
        float iw = fmaxf(rb0 - lt0, 0.f), ih = fmaxf(rb1 - lt1, 0.f);
        float inter = iw * ih;
        float areaI = (ix2 - ix1) * (iy2 - iy1);
        float iou = inter / (areaI + myArea - inter + 1e-6f);
        if (iou > IOU_THRv) skeep[t] = 0;
      }
    }
    __syncthreads();
  }

  if (t == 0) {
    float* dp = det + (long)img * 100 * 5;
    int* lp = outlab + img * 100;
    int cnt = 0;
    for (int i = 0; i < 1000 && cnt < 100; ++i) {
      if (skeep[i]) {
        dp[cnt * 5 + 0] = sx1[i]; dp[cnt * 5 + 1] = sy1[i];
        dp[cnt * 5 + 2] = sx2[i]; dp[cnt * 5 + 3] = sy2[i];
        dp[cnt * 5 + 4] = ssc[i];
        lp[cnt] = slb[i];
        ++cnt;
      }
    }
    for (; cnt < 100; ++cnt) {
      dp[cnt * 5 + 0] = 0.f; dp[cnt * 5 + 1] = 0.f; dp[cnt * 5 + 2] = 0.f;
      dp[cnt * 5 + 3] = 0.f; dp[cnt * 5 + 4] = 0.f;
      lp[cnt] = 0;
    }
  }
}

// ------------------------------ host launch --------------------------------
extern "C" void kernel_launch(void* const* d_in, const int* in_sizes, int n_in,
                              void* d_out, int out_size, void* d_ws, size_t ws_size,
                              hipStream_t stream) {
  (void)in_sizes; (void)n_in; (void)out_size; (void)ws_size;
  static const int IN_CH[3]  = {512, 256, 128};
  static const int OUT_CH[3] = {1024, 512, 256};
  static const int FHW[3]    = {19, 38, 76};
  static const int STR[3]    = {32, 16, 8};
  static const float AW[3][3] = {{116.f, 156.f, 373.f}, {30.f, 62.f, 59.f}, {10.f, 16.f, 33.f}};
  static const float AH[3][3] = {{90.f, 198.f, 326.f}, {61.f, 45.f, 119.f}, {13.f, 30.f, 23.f}};
  static const int COFF[3]   = {0, 1083, 5415};

  char* ws = (char*)d_ws;
  size_t off = 0;
  auto alloc = [&](size_t bytes) -> char* {
    size_t r = (off + 255) & ~(size_t)255;
    off = r + bytes;
    return ws + r;
  };
  bf16*  wbf   = (bf16*) alloc((size_t)1024 * 4608 * 2);     // max Cout*K bf16
  float* biasb = (float*)alloc((size_t)1024 * 4);
  bf16*  wpbf  = (bf16*) alloc((size_t)256 * 1024 * 2);
  bf16*  ybuf  = (bf16*) alloc((size_t)256 * 92416 * 2);     // max N*Cout bf16
  float* pmb   = (float*)alloc((size_t)92416 * 256 * 4);     // max N*256 f32
  float* boxes = (float*)alloc((size_t)16 * NTOT * 4 * 4);
  float* confb = (float*)alloc((size_t)16 * NTOT * 4);
  float* clsb  = (float*)alloc((size_t)16 * NTOT * 4);
  int*   labb  = (int*)  alloc((size_t)16 * NTOT * 4);
  unsigned long long* keys = (unsigned long long*)alloc((size_t)16 * SORTN * 8);

  for (int L = 0; L < 3; ++L) {
    const float* x     = (const float*)d_in[L];
    const float* wb    = (const float*)d_in[3 + 7 * L];
    const float* gamma = (const float*)d_in[4 + 7 * L];
    const float* beta  = (const float*)d_in[5 + 7 * L];
    const float* mean  = (const float*)d_in[6 + 7 * L];
    const float* var   = (const float*)d_in[7 + 7 * L];
    const float* wp    = (const float*)d_in[8 + 7 * L];
    const float* bp    = (const float*)d_in[9 + 7 * L];
    const int Cin = IN_CH[L], Cout = OUT_CH[L], H = FHW[L], W = FHW[L];
    const int HW = H * W, N = 16 * HW, K = Cin * 9;

    prep_wb_k<<<dim3((unsigned)(((long)Cout * K + 255) / 256)), dim3(256), 0, stream>>>(
        wb, gamma, var, wbf, Cout, Cin);
    prep_bias_k<<<dim3((Cout + 255) / 256), dim3(256), 0, stream>>>(
        gamma, beta, mean, var, biasb, Cout);
    prep_wp_k<<<dim3((256 * Cout + 255) / 256), dim3(256), 0, stream>>>(wp, wpbf, Cout);

    dim3 gConv((N + 63) / 64, Cout / 128);
    conv3x3_wmma_k<<<gConv, dim3(256), 0, stream>>>(x, wbf, biasb, ybuf, Cin, H, W, N, Cout);

    dim3 gPred((N + 63) / 64, 2);
    pred_wmma_k<<<gPred, dim3(256), 0, stream>>>(ybuf, wpbf, bp, pmb, N, Cout);

    decode_k<<<dim3((unsigned)(((long)N * 3 + 255) / 256)), dim3(256), 0, stream>>>(
        pmb, boxes, confb, clsb, labb, H, W, N, STR[L],
        AW[L][0], AH[L][0], AW[L][1], AH[L][1], AW[L][2], AH[L][2], COFF[L]);
  }

  sort_init_k<<<dim3(16 * SORTN / 256), dim3(256), 0, stream>>>(confb, keys);
  for (int k = 2; k <= SORTN; k <<= 1)
    for (int j = k >> 1; j > 0; j >>= 1)
      bitonic_k<<<dim3(16 * SORTN / 256), dim3(256), 0, stream>>>(keys, k, j);

  float* det = (float*)d_out;
  int* outlab = (int*)(det + 16 * 100 * 5);
  nms_k<<<dim3(16), dim3(1024), 0, stream>>>(keys, boxes, clsb, labb, det, outlab);
}
